// DynamicGraphAttentionBlock_24343874633920
// MI455X (gfx1250) — matrix operand order repaired and checked by reading, not compiled, not run on hardware
//
#include <hip/hip_runtime.h>
#include <hip/hip_bf16.h>

typedef __attribute__((ext_vector_type(16))) __bf16 v16bf;
typedef __attribute__((ext_vector_type(8)))  __bf16 v8bf;
typedef __attribute__((ext_vector_type(4)))  __bf16 v4bf;
typedef __attribute__((ext_vector_type(8)))  float  v8f;
typedef int v4i_gcc __attribute__((vector_size(16)));

#define B_DIM   8
#define KPTS    4096
#define CDIM    128
#define KNN_K   16
#define NROWS   (B_DIM * KPTS)      // 32768
#define NCOLS   512                 // S_a | N_a | S_f | N_f
#define LEAKY   0.2f

// ---------------------------------------------------------------------------
// Kernel 1: LayerNorm + positional embedding -> xp (bf16, row-major [NROWS,128])
// one wave per row, 4 channels per lane
// ---------------------------------------------------------------------------
__global__ void ln_pos_kernel(const float* __restrict__ x,
                              const float* __restrict__ pos,
                              const float* __restrict__ gamma,
                              const float* __restrict__ beta,
                              __bf16* __restrict__ xp) {
    const int wave = threadIdx.x >> 5;
    const int lane = threadIdx.x & 31;
    const int row  = blockIdx.x * 8 + wave;
    if (row >= NROWS) return;
    const int c0 = lane * 4;

    float v[4];
    float s = 0.f, sq = 0.f;
#pragma unroll
    for (int e = 0; e < 4; ++e) {
        v[e] = x[(size_t)row * CDIM + c0 + e];
        s  += v[e];
        sq += v[e] * v[e];
    }
#pragma unroll
    for (int off = 16; off > 0; off >>= 1) {
        s  += __shfl_xor(s,  off, 32);
        sq += __shfl_xor(sq, off, 32);
    }
    const float mu  = s * (1.0f / CDIM);
    const float var = sq * (1.0f / CDIM) - mu * mu;
    const float rs  = rsqrtf(var + 1e-5f);

    v4bf out;
#pragma unroll
    for (int e = 0; e < 4; ++e) {
        const int c = c0 + e;
        float xn = (v[e] - mu) * rs * gamma[c] + beta[c] + pos[(size_t)row * CDIM + c];
        out[e] = (__bf16)xn;
    }
    *(v4bf*)(xp + (size_t)row * CDIM + c0) = out;
}

// ---------------------------------------------------------------------------
// Kernel 2: pack concatenated weights [128,512] (f32 sources) into bf16
// WMMA-B fragment order:  frag(ntile,kstep) -> 32 lanes x 16 contiguous bf16
//   lane<16 : col = n0+lane,    elems 0-7 = K k0+0..7,  elems 8-15 = K k0+16..23
//   lane>=16: col = n0+lane-16, elems 0-7 = K k0+8..15, elems 8-15 = K k0+24..31
// ---------------------------------------------------------------------------
__global__ void pack_w_kernel(const float* __restrict__ W_a1,   // [256,128]
                              const float* __restrict__ W_f1,   // [256,128]
                              __bf16* __restrict__ packedB) {
    const int t = blockIdx.x * blockDim.x + threadIdx.x;   // 0 .. 65535
    if (t >= 32 * 4 * 32 * 16) return;
    const int kk    = t & 15;
    const int lane  = (t >> 4) & 31;
    const int kstep = (t >> 9) & 3;
    const int ntile = t >> 11;

    const int k = kstep * 32 + ((lane < 16) ? 0 : 8) + (kk & 7) + ((kk >> 3) << 4);    // 0..127
    const int n = ntile * 16 + (lane & 15);                    // 0..511

    float w;
    if (n < 128)       w = W_a1[(size_t)k * 128 + n];
    else if (n < 256)  w = W_a1[(size_t)(128 + k) * 128 + (n - 128)];
    else if (n < 384)  w = W_f1[(size_t)k * 128 + (n - 256)]
                         - W_f1[(size_t)(128 + k) * 128 + (n - 256)];
    else               w = W_f1[(size_t)(128 + k) * 128 + (n - 384)];
    packedB[t] = (__bf16)w;
}

// ---------------------------------------------------------------------------
// Kernel 3: GEMM  Y[32768,512] = xp[32768,128] @ Wcat[128,512]  (bf16 -> f32)
// block = 8 waves; block owns one 16-row M tile; wave w covers ntiles 4w..4w+3
// A tile (16x128 bf16 = 4KB) staged to LDS once per block via CDNA5 async
// global->LDS copy (ASYNCcnt); B fragments streamed from L2.
// ---------------------------------------------------------------------------
__global__ void gemm_kernel(const __bf16* __restrict__ xp,
                            const __bf16* __restrict__ packedB,
                            float* __restrict__ Y) {
    __shared__ __bf16 Atile[16 * CDIM];          // 4 KB
    const int wave = threadIdx.x >> 5;
    const int lane = threadIdx.x & 31;
    const int m0   = blockIdx.x * 16;
    const int nt0  = wave * 4;

    // --- stage A tile: 256 threads x 16B = 4KB ---
    {
        const __bf16* src = xp + (size_t)m0 * CDIM + threadIdx.x * 8;
        __bf16*       dst = &Atile[threadIdx.x * 8];
#if defined(__gfx1250__) && __has_builtin(__builtin_amdgcn_global_load_async_to_lds_b128)
        __builtin_amdgcn_global_load_async_to_lds_b128(
            (__attribute__((address_space(1))) v4i_gcc*)src,
            (__attribute__((address_space(3))) v4i_gcc*)dst,
            /*offset=*/0, /*cpol=*/0);
        asm volatile("s_wait_asynccnt 0x0" ::: "memory");
#else
        *(v8bf*)dst = *(const v8bf*)src;
#endif
    }
    __syncthreads();

    // A fragment addressing (16-bit A 16x32 layout, ISA 7.12.2):
    //   lane<16 : row m0+lane,    elems 0-7 = K k0+0..7,  elems 8-15 = K k0+16..23
    //   lane>=16: row m0+lane-16, elems 0-7 = K k0+8..15, elems 8-15 = K k0+24..31
    const int kbias = (lane < 16) ? 0 : 8;
    const __bf16* arowp = &Atile[(lane & 15) * CDIM];

    v8f acc[4];
#pragma unroll
    for (int nt = 0; nt < 4; ++nt) acc[nt] = (v8f)(0.f);

#pragma unroll
    for (int kstep = 0; kstep < 4; ++kstep) {
        const int k0 = kstep * 32;
        const v8bf alo = *(const v8bf*)(arowp + k0 + kbias);        // ds_load_b128
        const v8bf ahi = *(const v8bf*)(arowp + k0 + kbias + 16);   // ds_load_b128
        v16bf a;
#pragma unroll
        for (int e = 0; e < 8; ++e) { a[e] = alo[e]; a[e + 8] = ahi[e]; }

#pragma unroll
        for (int nt = 0; nt < 4; ++nt) {
            const int ntile = nt0 + nt;
            const v16bf b = *(const v16bf*)(packedB +
                               ((size_t)(ntile * 4 + kstep) * 32 + lane) * 16);
            acc[nt] = __builtin_amdgcn_wmma_f32_16x16x32_bf16(
                false, a, false, b, (short)0, acc[nt], false, false);
        }
    }

    // C layout: VGPR v, lane L: row = m0 + v + (L<16?0:8), col = n0 + (L&15)
    const int ccol  = lane & 15;
    const int rbase = (lane < 16) ? 0 : 8;
#pragma unroll
    for (int nt = 0; nt < 4; ++nt) {
        const int n0 = (nt0 + nt) * 16;
#pragma unroll
        for (int v = 0; v < 8; ++v) {
            Y[(size_t)(m0 + rbase + v) * NCOLS + n0 + ccol] = acc[nt][v];
        }
    }
}

// ---------------------------------------------------------------------------
// Kernel 4: KNN over rel_pos_coords [8,4096,3]; block = 256 queries of one
// batch; candidates tiled through LDS; branchless register top-16.
// ---------------------------------------------------------------------------
__global__ void knn_kernel(const float* __restrict__ coords,
                           int* __restrict__ knn_idx) {
    __shared__ float4 cpos[256];
    const int b   = blockIdx.x >> 4;
    const int q   = (blockIdx.x & 15) * 256 + threadIdx.x;   // 0..4095
    const float* cb = coords + (size_t)b * KPTS * 3;

    const float qx = cb[(size_t)q * 3 + 0];
    const float qy = cb[(size_t)q * 3 + 1];
    const float qz = cb[(size_t)q * 3 + 2];
    const float qsq = qx * qx + qy * qy + qz * qz;

    float dist16[KNN_K];
    int   idx16[KNN_K];
#pragma unroll
    for (int p = 0; p < KNN_K; ++p) { dist16[p] = 1e30f; idx16[p] = 0; }

    for (int tile = 0; tile < 16; ++tile) {
        const int cg = tile * 256 + threadIdx.x;
        {
            const float cx = cb[(size_t)cg * 3 + 0];
            const float cy = cb[(size_t)cg * 3 + 1];
            const float cz = cb[(size_t)cg * 3 + 2];
            cpos[threadIdx.x] = make_float4(cx, cy, cz, cx * cx + cy * cy + cz * cz);
        }
        __syncthreads();
#pragma unroll 4
        for (int cc = 0; cc < 256; ++cc) {
            const int cand = tile * 256 + cc;
            const float4 c = cpos[cc];
            float d = qsq + c.w - 2.0f * (qx * c.x + qy * c.y + qz * c.z);
            if (cand == q) d = 1e9f;                      // exclude self
            if (d < dist16[KNN_K - 1]) {
                float nd = d; int ni = cand;
#pragma unroll
                for (int p = 0; p < KNN_K; ++p) {         // bubble insert
                    if (nd < dist16[p]) {
                        float td = dist16[p]; dist16[p] = nd; nd = td;
                        int   ti = idx16[p];  idx16[p]  = ni; ni = ti;
                    }
                }
            }
        }
        __syncthreads();
    }

    const size_t base = ((size_t)b * KPTS + q) * KNN_K;
#pragma unroll
    for (int p = 0; p < KNN_K; ++p) knn_idx[base + p] = idx16[p];
}

// ---------------------------------------------------------------------------
// Kernel 5: per-point edge fusion. wave = 1 point, 4 channels per lane.
// pass1: logits_j = dot(leaky(S_a+N_a+b_a1), W_a2)+b_a2 -> softmax
//        (+ prefetch of the pass-2 N_f cachelines)
// pass2: agg = sum_j attn_j * leaky(S_f+N_f+b_f1)
// ---------------------------------------------------------------------------
__global__ void edge_kernel(const float* __restrict__ Y,        // [NROWS,512]
                            const int*   __restrict__ knn_idx,  // [NROWS,16]
                            const float* __restrict__ b_a1,
                            const float* __restrict__ W_a2,     // [128,1]
                            const float* __restrict__ b_a2,     // [1]
                            const float* __restrict__ b_f1,
                            float* __restrict__ out) {          // [NROWS,128]
    const int wave = threadIdx.x >> 5;
    const int lane = threadIdx.x & 31;
    const int i    = blockIdx.x * 8 + wave;
    if (i >= NROWS) return;
    const int  batch = i >> 12;
    const int  c0    = lane * 4;

    float sa[4], sf[4], ba1[4], bf1[4], wa2[4];
    const float* yrow = Y + (size_t)i * NCOLS;
#pragma unroll
    for (int e = 0; e < 4; ++e) {
        sa[e]  = yrow[c0 + e];             // S_a
        sf[e]  = yrow[256 + c0 + e];       // S_f
        ba1[e] = b_a1[c0 + e];
        bf1[e] = b_f1[c0 + e];
        wa2[e] = W_a2[c0 + e];
    }
    const float ba2 = b_a2[0];

    int nrow[KNN_K];
#pragma unroll
    for (int j = 0; j < KNN_K; ++j)
        nrow[j] = (batch << 12) + knn_idx[(size_t)i * KNN_K + j];

    // pass 1: logits
    float logit[KNN_K];
#pragma unroll
    for (int j = 0; j < KNN_K; ++j) {
        const float* nr = Y + (size_t)nrow[j] * NCOLS + 128;   // N_a
        // hide pass-2 latency: prefetch this neighbor's N_f region
        __builtin_prefetch(Y + (size_t)nrow[j] * NCOLS + 384 + c0, 0, 1);
        float part = 0.f;
#pragma unroll
        for (int e = 0; e < 4; ++e) {
            float h = sa[e] + nr[c0 + e] + ba1[e];
            h = (h > 0.f) ? h : LEAKY * h;
            part += h * wa2[e];
        }
#pragma unroll
        for (int off = 16; off > 0; off >>= 1)
            part += __shfl_xor(part, off, 32);
        logit[j] = part + ba2;
    }

    float m = logit[0];
#pragma unroll
    for (int j = 1; j < KNN_K; ++j) m = fmaxf(m, logit[j]);
    float z = 0.f;
#pragma unroll
    for (int j = 0; j < KNN_K; ++j) { logit[j] = __expf(logit[j] - m); z += logit[j]; }
    const float invz = 1.0f / z;

    // pass 2: weighted aggregation
    float agg[4] = {0.f, 0.f, 0.f, 0.f};
#pragma unroll
    for (int j = 0; j < KNN_K; ++j) {
        const float w = logit[j] * invz;
        const float* nr = Y + (size_t)nrow[j] * NCOLS + 384;   // N_f
#pragma unroll
        for (int e = 0; e < 4; ++e) {
            float t = sf[e] + nr[c0 + e] + bf1[e];
            t = (t > 0.f) ? t : LEAKY * t;
            agg[e] += w * t;
        }
    }
#pragma unroll
    for (int e = 0; e < 4; ++e) out[(size_t)i * CDIM + c0 + e] = agg[e];
}

// ---------------------------------------------------------------------------
extern "C" void kernel_launch(void* const* d_in, const int* in_sizes, int n_in,
                              void* d_out, int out_size, void* d_ws, size_t ws_size,
                              hipStream_t stream) {
    const float* x     = (const float*)d_in[0];
    const float* pos   = (const float*)d_in[1];
    const float* rpc   = (const float*)d_in[2];
    const float* gamma = (const float*)d_in[3];
    const float* beta  = (const float*)d_in[4];
    const float* W_a1  = (const float*)d_in[5];
    const float* b_a1  = (const float*)d_in[6];
    const float* W_a2  = (const float*)d_in[7];
    const float* b_a2  = (const float*)d_in[8];
    const float* W_f1  = (const float*)d_in[9];
    const float* b_f1  = (const float*)d_in[10];
    float* out = (float*)d_out;

    // workspace layout (256B aligned chunks)
    char* ws = (char*)d_ws;
    size_t off = 0;
    __bf16* xp = (__bf16*)(ws + off);            off += (size_t)NROWS * CDIM * 2;      // 8 MB
    off = (off + 255) & ~(size_t)255;
    __bf16* packedB = (__bf16*)(ws + off);       off += (size_t)32 * 4 * 32 * 16 * 2;  // 128 KB
    off = (off + 255) & ~(size_t)255;
    float* Y = (float*)(ws + off);               off += (size_t)NROWS * NCOLS * 4;     // 64 MB
    off = (off + 255) & ~(size_t)255;
    int* knn_idx = (int*)(ws + off);             off += (size_t)NROWS * KNN_K * 4;     // 2 MB

    pack_w_kernel<<<256, 256, 0, stream>>>(W_a1, W_f1, packedB);
    ln_pos_kernel<<<NROWS / 8, 256, 0, stream>>>(x, pos, gamma, beta, xp);
    gemm_kernel<<<NROWS / 16, 256, 0, stream>>>(xp, packedB, Y);
    knn_kernel<<<B_DIM * 16, 256, 0, stream>>>(rpc, knn_idx);
    edge_kernel<<<NROWS / 8, 256, 0, stream>>>(Y, knn_idx, b_a1, W_a2, b_a2, b_f1, out);
}
